// ColightEncoder_90950227460700
// MI455X (gfx1250) — matrix operand order repaired and checked by reading, not compiled, not run on hardware
//
#include <hip/hip_runtime.h>
#include <math.h>

#define BB   32
#define AA   200
#define NNE  5
#define DIN  36
#define HD   128
#define HEADS 5
#define NHID 640            // HD*HEADS
#define ROWS (BB*AA)        // 6400
#define MPAD 1024           // per-batch padded rows (1000 -> 1024, multiple of 32)
#define KADJ 224            // 200 -> 224 (multiple of 32)

typedef _Float16 half8  __attribute__((ext_vector_type(8)));
typedef _Float16 half16 __attribute__((ext_vector_type(16)));
typedef float    float8 __attribute__((ext_vector_type(8)));

// ---------------------------------------------------------------------------
// Generic WMMA GEMM: C(MxN) = act(A(MxK) @ B(KxN) + bias)
// A: f16 row-major (lda); B passed transposed: BT is N x K row-major (ldb)
// so B-fragments are contiguous 32B loads. One wave -> 32x64 tile:
// 2 A-fragments x 4 B-fragments -> 8 WMMAs per 32-wide k-step (B reused 2x).
// Optional batch (grid.z) strides; optional f32 output (final layer -> d_out).
// M % 32 == 0, N % 64 == 0, K % 32 == 0, lda % 8 == 0, ldb % 16 == 0.
// ---------------------------------------------------------------------------
__global__ __launch_bounds__(256) void gemm_wmma(
    const _Float16* __restrict__ A, int lda, long sA,
    const _Float16* __restrict__ BT, int ldb, long sB,
    const float* __restrict__ bias,
    _Float16* __restrict__ C, int ldc, long sC,
    float* __restrict__ Cf,
    int M, int N, int K, int doRelu)
{
    const int lane  = threadIdx.x & 31;
    const int wave  = threadIdx.x >> 5;
    const int tilesN = N >> 6;
    const int t = blockIdx.x * 8 + wave;          // tile id (wave-uniform)
    const int tileM = t / tilesN;
    if ((tileM << 5) >= M) return;                // whole wave exits: EXEC stays all-1s
    const int colBase = (t % tilesN) << 6;

    const long z = blockIdx.z;
    A  += z * sA;
    BT += z * sB;

    const int r16 = lane & 15;
    const int hi  = lane >> 4;                    // half-wave select

    // A-fragment bases: rows tileM*32 + r16 (+16); per half-wave K offset 0 / 16
    const _Float16* arow0 = A + (long)((tileM << 5) + r16) * lda + hi * 16;
    const _Float16* arow1 = arow0 + (long)16 * lda;
    // B-fragment base: col = colBase + r16; per half-wave K offset 0 / 16
    const _Float16* bbase = BT + (long)(colBase + r16) * ldb + hi * 16;

    float8 acc[2][4] = {};

    for (int k0 = 0; k0 < K; k0 += 32) {
        // A 16x32 f16 fragments (contiguous layout: lanes<16 hold K {0..15}, lanes>=16 hold K {16..31})
        half16 a0, a1;
        {
            half8 lo0 = *(const half8*)(arow0 + k0);
            half8 hi0 = *(const half8*)(arow0 + k0 + 8);
            half8 lo1 = *(const half8*)(arow1 + k0);
            half8 hi1 = *(const half8*)(arow1 + k0 + 8);
#pragma unroll
            for (int i = 0; i < 8; ++i) {
                a0[i] = lo0[i]; a0[i + 8] = hi0[i];
                a1[i] = lo1[i]; a1[i + 8] = hi1[i];
            }
        }
#pragma unroll
        for (int f = 0; f < 4; ++f) {
            half16 b = *(const half16*)(bbase + (long)(f * 16) * ldb + k0);
            acc[0][f] = __builtin_amdgcn_wmma_f32_16x16x32_f16(
                false, a0, false, b, (short)0, acc[0][f], false, false);
            acc[1][f] = __builtin_amdgcn_wmma_f32_16x16x32_f16(
                false, a1, false, b, (short)0, acc[1][f], false, false);
        }
    }

#pragma unroll
    for (int m = 0; m < 2; ++m) {
        const long row0 = (long)(tileM << 5) + m * 16;
#pragma unroll
        for (int f = 0; f < 4; ++f) {
            const int col = colBase + f * 16 + r16;
            const float bv = bias ? bias[col] : 0.0f;
#pragma unroll
            for (int j = 0; j < 8; ++j) {
                const long row = row0 + j + hi * 8;   // C layout: vgpr j -> M=j (+8 hi half)
                float v = acc[m][f][j] + bv;
                if (doRelu && v < 0.0f) v = 0.0f;
                if (C)  C [z * sC + row * ldc + col] = (_Float16)v;
                if (Cf) Cf[row * (long)ldc + col]    = v;
            }
        }
    }
}

// --- conversion / transpose helpers -----------------------------------------
__global__ void conv_wT(const float* __restrict__ W, _Float16* __restrict__ WT,
                        int K, int N, int ldk) {
    int idx = blockIdx.x * blockDim.x + threadIdx.x;
    if (idx >= N * ldk) return;
    int n = idx / ldk, k = idx % ldk;
    WT[idx] = (k < K) ? (_Float16)W[(long)k * N + n] : (_Float16)0.0f;
}

__global__ void conv_feat(const float* __restrict__ X, _Float16* __restrict__ Xp) {
    int idx = blockIdx.x * blockDim.x + threadIdx.x;
    if (idx >= ROWS * 64) return;
    int r = idx >> 6, k = idx & 63;
    Xp[idx] = (k < DIN) ? (_Float16)X[(long)r * DIN + k] : (_Float16)0.0f;
}

__global__ void conv_adj(const float* __restrict__ adj, _Float16* __restrict__ adjP) {
    long idx = (long)blockIdx.x * blockDim.x + threadIdx.x;
    if (idx >= (long)BB * MPAD * KADJ) return;
    int k = (int)(idx % KADJ);
    long tmp = idx / KADJ;
    int r = (int)(tmp % MPAD);
    int b = (int)(tmp / MPAD);
    _Float16 v = (_Float16)0.0f;
    if (r < AA * NNE && k < AA) v = (_Float16)adj[((long)b * (AA * NNE) + r) * AA + k];
    adjP[idx] = v;
}

__global__ void trans_h(const _Float16* __restrict__ h, _Float16* __restrict__ hT) {
    int idx = blockIdx.x * blockDim.x + threadIdx.x;     // BB*HD*KADJ
    if (idx >= BB * HD * KADJ) return;
    int k = idx % KADJ;
    int tmp = idx / KADJ;
    int n = tmp % HD;
    int b = tmp / HD;
    hT[idx] = (k < AA) ? h[((long)b * AA + k) * HD + n] : (_Float16)0.0f;
}

// --- attention scores: s[hd][n] = <ah[row, ::, hd], nh[row,n, ::, hd]>, softmax over n
__global__ void attn_scores(const _Float16* __restrict__ ah,
                            const _Float16* __restrict__ nh,
                            float* __restrict__ att) {
    const int row = blockIdx.x;                 // 0..6399
    const int b = row / AA, a = row % AA;
    __shared__ float s[32];
    const int t = threadIdx.x;
    if (t < HEADS * NNE) {
        const int hd = t / NNE, n = t % NNE;
        const _Float16* ap = ah + (long)row * NHID;
        const _Float16* np = nh + ((long)b * MPAD + a * NNE + n) * NHID;
        float acc = 0.0f;
        for (int d = 0; d < HD; ++d)
            acc += (float)ap[d * HEADS + hd] * (float)np[d * HEADS + hd];
        s[t] = acc;
    }
    __syncthreads();
    if (t < HEADS * NNE) {
        const int basei = (t / NNE) * NNE;
        float m = s[basei];
        for (int i = 1; i < NNE; ++i) m = fmaxf(m, s[basei + i]);
        float sum = 0.0f;
        for (int i = 0; i < NNE; ++i) sum += __expf(s[basei + i] - m);
        att[(long)row * 25 + t] = __expf(s[t] - m) / sum;
    }
}

// --- out[row,d] = mean_hd sum_n att[row,hd,n] * hh[row,n, d,hd]
__global__ void attn_combine(const float* __restrict__ att,
                             const _Float16* __restrict__ hh,
                             _Float16* __restrict__ outp) {
    int idx = blockIdx.x * blockDim.x + threadIdx.x;     // ROWS*HD
    if (idx >= ROWS * HD) return;
    const int d = idx & (HD - 1);
    const int row = idx >> 7;
    const int b = row / AA, a = row % AA;
    const float* at = att + (long)row * 25;
    float acc = 0.0f;
#pragma unroll
    for (int n = 0; n < NNE; ++n) {
        const _Float16* hp = hh + ((long)b * MPAD + a * NNE + n) * NHID + d * HEADS;
#pragma unroll
        for (int hd = 0; hd < HEADS; ++hd)
            acc += at[hd * NNE + n] * (float)hp[hd];
    }
    outp[idx] = (_Float16)(acc * 0.2f);
}

// ---------------------------------------------------------------------------
extern "C" void kernel_launch(void* const* d_in, const int* in_sizes, int n_in,
                              void* d_out, int out_size, void* d_ws, size_t ws_size,
                              hipStream_t stream) {
    (void)in_sizes; (void)n_in; (void)out_size; (void)ws_size;

    const float* features = (const float*)d_in[0];
    const float* adj      = (const float*)d_in[1];
    const float* W1 = (const float*)d_in[2];
    const float* b1 = (const float*)d_in[3];
    const float* W2 = (const float*)d_in[4];
    const float* b2 = (const float*)d_in[5];
    const float* Wa[2] = {(const float*)d_in[6],  (const float*)d_in[14]};
    const float* ba[2] = {(const float*)d_in[7],  (const float*)d_in[15]};
    const float* Wn[2] = {(const float*)d_in[8],  (const float*)d_in[16]};
    const float* bn[2] = {(const float*)d_in[9],  (const float*)d_in[17]};
    const float* Wh[2] = {(const float*)d_in[10], (const float*)d_in[18]};
    const float* bh[2] = {(const float*)d_in[11], (const float*)d_in[19]};
    const float* Wo[2] = {(const float*)d_in[12], (const float*)d_in[20]};
    const float* bo[2] = {(const float*)d_in[13], (const float*)d_in[21]};

    char* base = (char*)d_ws;
    size_t off = 0;
    auto alloc = [&](size_t bytes) -> char* {
        off = (off + 255) & ~(size_t)255;
        char* p = base + off;
        off += bytes;
        return p;
    };

    _Float16* W1T = (_Float16*)alloc((size_t)HD * 64 * 2);
    _Float16* W2T = (_Float16*)alloc((size_t)HD * HD * 2);
    _Float16 *WaT[2], *WnT[2], *WhT[2], *WoT[2];
    for (int p = 0; p < 2; ++p) {
        WaT[p] = (_Float16*)alloc((size_t)NHID * HD * 2);
        WnT[p] = (_Float16*)alloc((size_t)NHID * HD * 2);
        WhT[p] = (_Float16*)alloc((size_t)NHID * HD * 2);
        WoT[p] = (_Float16*)alloc((size_t)HD * HD * 2);
    }
    _Float16* Xp   = (_Float16*)alloc((size_t)ROWS * 64 * 2);
    _Float16* h0   = (_Float16*)alloc((size_t)ROWS * HD * 2);
    _Float16* hbuf = (_Float16*)alloc((size_t)ROWS * HD * 2);
    _Float16* hT   = (_Float16*)alloc((size_t)BB * HD * KADJ * 2);
    _Float16* adjP = (_Float16*)alloc((size_t)BB * MPAD * KADJ * 2);
    _Float16* nei  = (_Float16*)alloc((size_t)BB * MPAD * HD * 2);
    _Float16* ahB  = (_Float16*)alloc((size_t)ROWS * NHID * 2);
    _Float16* P    = (_Float16*)alloc((size_t)BB * MPAD * NHID * 2);   // nh then hh
    float*    att  = (float*)   alloc((size_t)ROWS * 25 * 4);
    _Float16* outB = (_Float16*)alloc((size_t)ROWS * HD * 2);

    auto gemm = [&](const _Float16* A, int lda, long sA,
                    const _Float16* BT, int ldb, long sB,
                    const float* bias, _Float16* C, int ldc, long sC,
                    float* Cf, int M, int N, int K, int relu, int batches) {
        int tiles = (M / 32) * (N / 64);
        dim3 grid((tiles + 7) / 8, 1, batches);
        gemm_wmma<<<grid, 256, 0, stream>>>(A, lda, sA, BT, ldb, sB, bias,
                                            C, ldc, sC, Cf, M, N, K, relu);
    };

    // ---- one-time conversions ----
    conv_wT<<<(HD * 64 + 255) / 256, 256, 0, stream>>>(W1, W1T, DIN, HD, 64);
    conv_wT<<<(HD * HD + 255) / 256, 256, 0, stream>>>(W2, W2T, HD, HD, HD);
    for (int p = 0; p < 2; ++p) {
        conv_wT<<<(NHID * HD + 255) / 256, 256, 0, stream>>>(Wa[p], WaT[p], HD, NHID, HD);
        conv_wT<<<(NHID * HD + 255) / 256, 256, 0, stream>>>(Wn[p], WnT[p], HD, NHID, HD);
        conv_wT<<<(NHID * HD + 255) / 256, 256, 0, stream>>>(Wh[p], WhT[p], HD, NHID, HD);
        conv_wT<<<(HD * HD + 255) / 256, 256, 0, stream>>>(Wo[p], WoT[p], HD, HD, HD);
    }
    conv_feat<<<(ROWS * 64 + 255) / 256, 256, 0, stream>>>(features, Xp);
    {
        long tot = (long)BB * MPAD * KADJ;
        conv_adj<<<(int)((tot + 255) / 256), 256, 0, stream>>>(adj, adjP);
    }

    // ---- MLP: h = relu(relu(X@W1+b1)@W2+b2) ----
    gemm(Xp, 64, 0, W1T, 64, 0, b1, h0, HD, 0, nullptr, ROWS, HD, 64, 1, 1);
    gemm(h0, HD, 0, W2T, HD, 0, b2, hbuf, HD, 0, nullptr, ROWS, HD, HD, 1, 1);

    // ---- two attention blocks ----
    for (int p = 0; p < 2; ++p) {
        trans_h<<<(BB * HD * KADJ + 255) / 256, 256, 0, stream>>>(hbuf, hT);

        // nei[b] = adj[b] @ h[b]   (batched over grid.z)
        gemm(adjP, KADJ, (long)MPAD * KADJ,
             hT,   KADJ, (long)HD * KADJ,
             nullptr, nei, HD, (long)MPAD * HD, nullptr,
             MPAD, HD, KADJ, 0, BB);

        // ah = relu(h @ Wa + ba)
        gemm(hbuf, HD, 0, WaT[p], HD, 0, ba[p], ahB, NHID, 0, nullptr,
             ROWS, NHID, HD, 1, 1);

        // nh = relu(nei @ Wn + bn)
        gemm(nei, HD, 0, WnT[p], HD, 0, bn[p], P, NHID, 0, nullptr,
             BB * MPAD, NHID, HD, 1, 1);

        attn_scores<<<ROWS, 32, 0, stream>>>(ahB, P, att);

        // hh = relu(nei @ Wh + bh)  (reuse P)
        gemm(nei, HD, 0, WhT[p], HD, 0, bh[p], P, NHID, 0, nullptr,
             BB * MPAD, NHID, HD, 1, 1);

        attn_combine<<<(ROWS * HD + 255) / 256, 256, 0, stream>>>(att, P, outB);

        // y = relu(out @ Wo + bo)
        if (p == 0) {
            gemm(outB, HD, 0, WoT[p], HD, 0, bo[p], hbuf, HD, 0, nullptr,
                 ROWS, HD, HD, 1, 1);
        } else {
            gemm(outB, HD, 0, WoT[p], HD, 0, bo[p], nullptr, HD, 0,
                 (float*)d_out, ROWS, HD, HD, 1, 1);
        }
    }
}
